// BatchedTripletLayer_87187836109385
// MI455X (gfx1250) — matrix operand order repaired and checked by reading, not compiled, not run on hardware
//
#include <hip/hip_runtime.h>
#include <hip/hip_bf16.h>
#include <math.h>

// ---------------- types / helpers ----------------
typedef __bf16 bf16_t;
typedef __attribute__((ext_vector_type(16))) __bf16 v16bf;
typedef __attribute__((ext_vector_type(8)))  __bf16 v8bf;
typedef __attribute__((ext_vector_type(8)))  float  v8f;

#define DEV static __device__ __forceinline__

DEV float  bf2f(bf16_t x) { return (float)x; }
DEV bf16_t f2bf(float x)  { return (bf16_t)x; }
DEV float  gelu_exact(float x) { return 0.5f * x * (1.0f + erff(x * 0.7071067811865475f)); }
DEV float  sigm(float x) { return 1.0f / (1.0f + __expf(-x)); }
DEV float  wave_max(float v) { for (int o = 16; o > 0; o >>= 1) v = fmaxf(v, __shfl_xor(v, o, 32)); return v; }
DEV float  wave_sum(float v) { for (int o = 16; o > 0; o >>= 1) v += __shfl_xor(v, o, 32); return v; }

// A fragment (16x32 bf16 tile, row-major LDS, stride sA multiple of 8):
// lane (m = lane&15, hl = lane>>4) holds A[m][8*hl..8*hl+8) ++ A[m][16+8*hl..+8)  -> 2x ds_load_b128
DEV v16bf load_a_frag(const bf16_t* A, int sA, int lane)
{
    const int hl = lane >> 4, m = lane & 15;
    const v8bf lo = *(const v8bf*)(A + m * sA + 8 * hl);
    const v8bf hi = *(const v8bf*)(A + m * sA + 16 + 8 * hl);
    return __builtin_shufflevector(lo, hi, 0, 1, 2, 3, 4, 5, 6, 7, 8, 9, 10, 11, 12, 13, 14, 15);
}

// B fragment from TRANSPOSED tile (Bt[n][k], 16 rows x 32 K, stride sBt multiple of 8):
// lane (n = lane&15, hl) holds Bt[n][8*hl..8*hl+8) ++ Bt[n][16+8*hl..+8)  -> 2x ds_load_b128
DEV v16bf load_bt_frag(const bf16_t* Bt, int sBt, int lane)
{
    const int hl = lane >> 4, n = lane & 15;
    const v8bf lo = *(const v8bf*)(Bt + n * sBt + 8 * hl);
    const v8bf hi = *(const v8bf*)(Bt + n * sBt + 16 + 8 * hl);
    return __builtin_shufflevector(lo, hi, 0, 1, 2, 3, 4, 5, 6, 7, 8, 9, 10, 11, 12, 13, 14, 15);
}

// acc += A(16x32) * B(32x16);  A row-major LDS, Bt = transposed B tile (n-major)
DEV v8f wmma_tile(const bf16_t* A, int sA, const bf16_t* Bt, int sBt, v8f acc, int lane)
{
    const v16bf av = load_a_frag(A, sA, lane);
    const v16bf bv = load_bt_frag(Bt, sBt, lane);
    return __builtin_amdgcn_wmma_f32_16x16x32_bf16(false, av, false, bv, (short)0, acc, false, false);
}

DEV void copy_bf16_v8(bf16_t* dst, const bf16_t* src, int n_elems, int tid, int nthr)
{
    v8bf* d = (v8bf*)dst;
    const v8bf* s = (const v8bf*)src;
    const int n8 = n_elems >> 3;
    for (int i = tid; i < n8; i += nthr) d[i] = s[i];
}

// ---- CDNA5 async global->LDS (ASYNCcnt) ----
DEV unsigned lds_off32(const void* p) { return (unsigned)(unsigned long long)p; }  // low 32b of generic = LDS offset
DEV void async_ld16(unsigned lds_off, const bf16_t* g)
{
    asm volatile("global_load_async_to_lds_b128 %0, %1, off" :: "v"(lds_off), "v"(g) : "memory");
}
DEV void wait_async0()
{
    asm volatile("s_wait_asynccnt 0x0" ::: "memory");
}

#define ME 262144   // B*N*N
#define MN 1024     // B*N

// ---------------- weight convert: f32 [K][N] row-major -> bf16 TRANSPOSED [N][K] ----------------
__global__ __launch_bounds__(256) void k_convert(const float* __restrict__ src, bf16_t* __restrict__ dst,
                                                 int rows, int cols, int kstride, int coloff)
{
    int i = blockIdx.x * 256 + threadIdx.x;
    if (i < rows * cols) {
        int r = i / cols, c = i - r * cols;
        dst[(coloff + c) * kstride + r] = f2bf(src[i]);
    }
}

// ---------------- QKV: [1024,128] @ [128,384] -> f32 (Wqkv_t: [384][128]) ----------------
__global__ __launch_bounds__(256) void k_qkv(const float* __restrict__ H, const bf16_t* __restrict__ WqkvT,
                                             float* __restrict__ QKV)
{
    extern __shared__ char smem[];
    bf16_t* Xs = (bf16_t*)smem;            // 128x128 (row-major)
    bf16_t* Wc = (bf16_t*)(smem + 32768);  // 128n x 128k chunk (transposed)
    const int tid = threadIdx.x, lane = tid & 31, wv = tid >> 5;
    const long mbase = (long)blockIdx.x * 128;
    for (int i = tid; i < 128 * 128; i += 256) Xs[i] = f2bf(H[mbase * 128 + i]);
    for (int c3 = 0; c3 < 3; ++c3) {
        __syncthreads();
        copy_bf16_v8(Wc, WqkvT + (long)c3 * 128 * 128, 128 * 128, tid, 256);
        __syncthreads();
        const int mt = wv, n = lane & 15, mh = (lane >> 4) * 8;
        for (int nt = 0; nt < 8; ++nt) {
            v8f acc = {};
#pragma unroll
            for (int k0 = 0; k0 < 128; k0 += 32)
                acc = wmma_tile(&Xs[mt * 16 * 128 + k0], 128, &Wc[nt * 16 * 128 + k0], 128, acc, lane);
#pragma unroll
            for (int v = 0; v < 8; ++v)
                QKV[(mbase + mt * 16 + mh + v) * 384 + c3 * 128 + nt * 16 + n] = acc[v];
        }
    }
}

// ---------------- attention row: one block per (b,i), thread = j ----------------
__global__ __launch_bounds__(256) void k_attn_row(
    const float* __restrict__ E, const float* __restrict__ QKV,
    const float* __restrict__ Wbg, const float* __restrict__ Wa2e, const float* __restrict__ ba2e,
    const float* __restrict__ gea, const float* __restrict__ bea,
    const float* __restrict__ geffn, const float* __restrict__ beffn,
    float* __restrict__ Obuf, float* __restrict__ Eagg, bf16_t* __restrict__ Xln)
{
    __shared__ float Qi[128];
    __shared__ float Wbg_s[64 * 16];
    __shared__ float Wa_s[8 * 64];
    __shared__ float aux[64 * 5];   // ba2e | g_ea | b_ea | g_effn | b_effn
    __shared__ float osh[128];
    __shared__ float red[8 * 8];
    const int tid = threadIdx.x, lane = tid & 31, wv = tid >> 5;
    const int b = blockIdx.x >> 8, i = blockIdx.x & 255;
    const long rowQ = (long)(b * 256 + i);
    if (tid < 128) { Qi[tid] = QKV[rowQ * 384 + tid]; osh[tid] = 0.f; }
    for (int t = tid; t < 1024; t += 256) Wbg_s[t] = Wbg[t];
    for (int t = tid; t < 512; t += 256) Wa_s[t] = Wa2e[t];
    if (tid < 64) {
        aux[tid]        = ba2e[tid];
        aux[64 + tid]   = gea[tid];
        aux[128 + tid]  = bea[tid];
        aux[192 + tid]  = geffn[tid];
        aux[256 + tid]  = beffn[tid];
    }
    __syncthreads();

    const int j = tid;
    const long erow = ((long)(b * 256 + i) * 256 + j);
    float e[64];
#pragma unroll
    for (int c = 0; c < 64; ++c) e[c] = E[erow * 64 + c];

    float bias[8], gate[8];
#pragma unroll
    for (int h = 0; h < 8; ++h) { bias[h] = 0.f; gate[h] = 0.f; }
#pragma unroll
    for (int c = 0; c < 64; ++c) {
        float ec = e[c];
#pragma unroll
        for (int h = 0; h < 8; ++h) { bias[h] += ec * Wbg_s[c * 16 + h]; gate[h] += ec * Wbg_s[c * 16 + 8 + h]; }
    }

    const long rowK = (long)(b * 256 + j);
    float raw[8], z[8];
#pragma unroll
    for (int h = 0; h < 8; ++h) {
        float s = 0.f;
#pragma unroll
        for (int d = 0; d < 16; ++d) s += Qi[h * 16 + d] * QKV[rowK * 384 + 128 + h * 16 + d];
        raw[h] = s * 0.25f + bias[h];
        gate[h] = sigm(gate[h]);
        z[h] = raw[h] * gate[h];
    }

#pragma unroll
    for (int h = 0; h < 8; ++h) { float m = wave_max(z[h]); if (lane == 0) red[h * 8 + wv] = m; }
    __syncthreads();
    float Mx[8];
#pragma unroll
    for (int h = 0; h < 8; ++h) {
        float m = red[h * 8];
#pragma unroll
        for (int w = 1; w < 8; ++w) m = fmaxf(m, red[h * 8 + w]);
        Mx[h] = m;
    }
    __syncthreads();
    float ex[8];
#pragma unroll
    for (int h = 0; h < 8; ++h) { ex[h] = __expf(z[h] - Mx[h]); float s = wave_sum(ex[h]); if (lane == 0) red[h * 8 + wv] = s; }
    __syncthreads();
    float Sm[8];
#pragma unroll
    for (int h = 0; h < 8; ++h) {
        float s = red[h * 8];
#pragma unroll
        for (int w = 1; w < 8; ++w) s += red[h * 8 + w];
        Sm[h] = s;
    }

#pragma unroll
    for (int h = 0; h < 8; ++h) {
        float a = ex[h] / Sm[h];
#pragma unroll
        for (int d = 0; d < 16; ++d)
            atomicAdd(&osh[h * 16 + d], a * QKV[rowK * 384 + 256 + h * 16 + d]);
    }
    __syncthreads();
    if (tid < 128) Obuf[rowQ * 128 + tid] = osh[tid];

    float mean = 0.f;
#pragma unroll
    for (int c = 0; c < 64; ++c) mean += e[c];
    mean *= (1.0f / 64.0f);
    float var = 0.f;
#pragma unroll
    for (int c = 0; c < 64; ++c) { float d = e[c] - mean; var += d * d; }
    var *= (1.0f / 64.0f);
    float rs = rsqrtf(var + 1e-5f);
#pragma unroll
    for (int c = 0; c < 64; ++c) {
        float s = aux[c];
#pragma unroll
        for (int h = 0; h < 8; ++h) s += raw[h] * Wa_s[h * 64 + c];
        e[c] = (e[c] - mean) * rs * aux[64 + c] + aux[128 + c] + s;
    }
    float m2 = 0.f;
#pragma unroll
    for (int c = 0; c < 64; ++c) m2 += e[c];
    m2 *= (1.0f / 64.0f);
    float v2 = 0.f;
#pragma unroll
    for (int c = 0; c < 64; ++c) { float d = e[c] - m2; v2 += d * d; }
    v2 *= (1.0f / 64.0f);
    float rs2 = rsqrtf(v2 + 1e-5f);
#pragma unroll
    for (int c = 0; c < 64; ++c) {
        Eagg[erow * 64 + c] = e[c];
        Xln[erow * 64 + c] = f2bf((e[c] - m2) * rs2 * aux[192 + c] + aux[256 + c]);
    }
}

// ---------------- fused edge FFN (W1T: [256][64], W2T: [64][256]) ----------------
__global__ __launch_bounds__(256) void edge_ffn_kernel(
    const bf16_t* __restrict__ Xln, const float* __restrict__ Ebase,
    const bf16_t* __restrict__ W1T, const float* __restrict__ b1,
    const bf16_t* __restrict__ W2T, const float* __restrict__ b2,
    float* __restrict__ Out, bf16_t* __restrict__ LnOut,
    const float* __restrict__ gamma, const float* __restrict__ beta, int emit_ln)
{
    extern __shared__ char smem[];
    bf16_t* Xs  = (bf16_t*)smem;               // 128x64
    bf16_t* Ws  = (bf16_t*)(smem + 16384);     // 256n x 64k / 64n x 256k
    bf16_t* Hs  = (bf16_t*)(smem + 49152);     // 128x256 row-major
    float*  bsh = (float*)(smem + 114688);     // 256
    float*  Ys  = (float*)Hs;                  // 128x64 f32 overlay
    const int tid = threadIdx.x, lane = tid & 31, wv = tid >> 5;
    const long mbase = (long)blockIdx.x * 128;

    __builtin_prefetch(Ebase + mbase * 64 + tid * 32, 0, 1);   // residual stream -> global_prefetch

    copy_bf16_v8(Xs, Xln + mbase * 64, 128 * 64, tid, 256);
    copy_bf16_v8(Ws, W1T, 256 * 64, tid, 256);
    bsh[tid] = b1[tid];
    __syncthreads();

    // phase 1: H = gelu(X @ W1 + b1)
    {
        const int mt = wv, n = lane & 15, mh = (lane >> 4) * 8;
        for (int nt = 0; nt < 16; ++nt) {
            v8f acc = {};
            acc = wmma_tile(&Xs[mt * 16 * 64 + 0], 64, &Ws[nt * 16 * 64 + 0], 64, acc, lane);
            acc = wmma_tile(&Xs[mt * 16 * 64 + 32], 64, &Ws[nt * 16 * 64 + 32], 64, acc, lane);
#pragma unroll
            for (int v = 0; v < 8; ++v)
                Hs[(mt * 16 + mh + v) * 256 + nt * 16 + n] = f2bf(gelu_exact(acc[v] + bsh[nt * 16 + n]));
        }
    }
    __syncthreads();
    copy_bf16_v8(Ws, W2T, 64 * 256, tid, 256);
    if (tid < 64) bsh[tid] = b2[tid];
    __syncthreads();

    // phase 2: Y = H @ W2 + b2 ; Out = 2*Ebase + Y
    v8f acc2[4];
    {
        const int mt = wv;
#pragma unroll
        for (int nt = 0; nt < 4; ++nt) {
            v8f acc = {};
#pragma unroll
            for (int k0 = 0; k0 < 256; k0 += 32)
                acc = wmma_tile(&Hs[mt * 16 * 256 + k0], 256, &Ws[nt * 16 * 256 + k0], 256, acc, lane);
            acc2[nt] = acc;
        }
    }
    __syncthreads();  // all waves done reading Hs -> reuse as Ys
    {
        const int mt = wv, n = lane & 15, mh = (lane >> 4) * 8;
#pragma unroll
        for (int nt = 0; nt < 4; ++nt)
#pragma unroll
            for (int v = 0; v < 8; ++v) {
                int r = mt * 16 + mh + v, c = nt * 16 + n;
                float y = acc2[nt][v] + bsh[c] + 2.0f * Ebase[(mbase + r) * 64 + c];
                Ys[r * 64 + c] = y;
                Out[(mbase + r) * 64 + c] = y;
            }
    }
    __syncthreads();
    if (emit_ln && tid < 128) {
        float m = 0.f;
#pragma unroll
        for (int c = 0; c < 64; ++c) m += Ys[tid * 64 + c];
        m *= (1.0f / 64.0f);
        float var = 0.f;
#pragma unroll
        for (int c = 0; c < 64; ++c) { float d = Ys[tid * 64 + c] - m; var += d * d; }
        var *= (1.0f / 64.0f);
        float rs = rsqrtf(var + 1e-5f);
#pragma unroll
        for (int c = 0; c < 64; ++c)
            LnOut[(mbase + tid) * 64 + c] = f2bf((Ys[tid * 64 + c] - m) * rs * gamma[c] + beta[c]);
    }
}

// ---------------- triplet prep: v = En@Wv_in (WMMA, WvT [64][64]); v stored as v2[b][h][j][d][k] ----------------
__global__ __launch_bounds__(256) void k_trip_prep(
    const bf16_t* __restrict__ En, const bf16_t* __restrict__ WvT,
    const float* __restrict__ Wgb2, bf16_t* __restrict__ V2,
    float* __restrict__ gate2, float* __restrict__ bias2)
{
    __shared__ bf16_t Xs[128 * 64];
    __shared__ bf16_t Ws[64 * 64];
    __shared__ float  Wg[64 * 8];
    const int tid = threadIdx.x, lane = tid & 31, wv = tid >> 5;
    const long mbase = (long)blockIdx.x * 128;
    copy_bf16_v8(Xs, En + mbase * 64, 128 * 64, tid, 256);
    copy_bf16_v8(Ws, WvT, 64 * 64, tid, 256);
    for (int i = tid; i < 64 * 8; i += 256) Wg[i] = Wgb2[i];
    __syncthreads();
    {
        const int mt = wv, n = lane & 15, mh = (lane >> 4) * 8;
#pragma unroll
        for (int nt = 0; nt < 4; ++nt) {   // nt == head h, n == d
            v8f acc = {};
            acc = wmma_tile(&Xs[mt * 16 * 64 + 0], 64, &Ws[nt * 16 * 64 + 0], 64, acc, lane);
            acc = wmma_tile(&Xs[mt * 16 * 64 + 32], 64, &Ws[nt * 16 * 64 + 32], 64, acc, lane);
#pragma unroll
            for (int v = 0; v < 8; ++v) {
                const long r = mbase + mt * 16 + mh + v;        // edge row = ((b*256+j)*256+k)
                const int b = (int)(r >> 16), j = (int)((r >> 8) & 255), k = (int)(r & 255);
                V2[((((long)b * 4 + nt) * 256 + j) * 16 + n) * 256 + k] = f2bf(acc[v]);
            }
        }
    }
    if (tid < 128) {
        float s[8];
#pragma unroll
        for (int t = 0; t < 8; ++t) s[t] = 0.f;
        for (int c = 0; c < 64; ++c) {
            float xv = bf2f(Xs[tid * 64 + c]);
#pragma unroll
            for (int t = 0; t < 8; ++t) s[t] += xv * Wg[c * 8 + t];
        }
        const long r = mbase + tid;
#pragma unroll
        for (int t = 0; t < 4; ++t) { gate2[r * 4 + t] = sigm(s[t]); bias2[r * 4 + t] = s[4 + t]; }
    }
}

// ---------------- softmaxes -> bf16 [b][h][i][k] ----------------
__global__ __launch_bounds__(256) void k_softmax_in(const float* __restrict__ bias2,
                                                    const float* __restrict__ gate2, bf16_t* __restrict__ Win)
{
    __shared__ float red[4 * 8];
    const int tid = threadIdx.x, lane = tid & 31, wv = tid >> 5;
    const int b = blockIdx.x >> 8, i = blockIdx.x & 255, k = tid;
    const long base = (((long)b * 256 + i) * 256 + k) * 4;
    float z[4], g[4];
#pragma unroll
    for (int h = 0; h < 4; ++h) { z[h] = bias2[base + h]; g[h] = gate2[base + h]; }
#pragma unroll
    for (int h = 0; h < 4; ++h) { float m = wave_max(z[h]); if (lane == 0) red[h * 8 + wv] = m; }
    __syncthreads();
    float M[4];
#pragma unroll
    for (int h = 0; h < 4; ++h) {
        float m = red[h * 8];
#pragma unroll
        for (int w = 1; w < 8; ++w) m = fmaxf(m, red[h * 8 + w]);
        M[h] = m;
    }
    __syncthreads();
    float ex[4];
#pragma unroll
    for (int h = 0; h < 4; ++h) { ex[h] = __expf(z[h] - M[h]); float s = wave_sum(ex[h]); if (lane == 0) red[h * 8 + wv] = s; }
    __syncthreads();
#pragma unroll
    for (int h = 0; h < 4; ++h) {
        float s = red[h * 8];
#pragma unroll
        for (int w = 1; w < 8; ++w) s += red[h * 8 + w];
        Win[(((long)b * 4 + h) * 256 + i) * 256 + k] = f2bf(ex[h] / s * g[h]);
    }
}

__global__ __launch_bounds__(256) void k_softmax_out(const float* __restrict__ bias2,
                                                     const float* __restrict__ gate2, bf16_t* __restrict__ Wout)
{
    __shared__ float red[4 * 8];
    const int tid = threadIdx.x, lane = tid & 31, wv = tid >> 5;
    const int b = blockIdx.x >> 8, k = blockIdx.x & 255, i = tid;
    const long base = (((long)b * 256 + i) * 256 + k) * 4;
    float z[4], g[4];
#pragma unroll
    for (int h = 0; h < 4; ++h) { z[h] = bias2[base + h]; g[h] = gate2[base + h]; }
#pragma unroll
    for (int h = 0; h < 4; ++h) { float m = wave_max(z[h]); if (lane == 0) red[h * 8 + wv] = m; }
    __syncthreads();
    float M[4];
#pragma unroll
    for (int h = 0; h < 4; ++h) {
        float m = red[h * 8];
#pragma unroll
        for (int w = 1; w < 8; ++w) m = fmaxf(m, red[h * 8 + w]);
        M[h] = m;
    }
    __syncthreads();
    float ex[4];
#pragma unroll
    for (int h = 0; h < 4; ++h) { ex[h] = __expf(z[h] - M[h]); float s = wave_sum(ex[h]); if (lane == 0) red[h * 8 + wv] = s; }
    __syncthreads();
#pragma unroll
    for (int h = 0; h < 4; ++h) {
        float s = red[h * 8];
#pragma unroll
        for (int w = 1; w < 8; ++w) s += red[h * 8 + w];
        Wout[(((long)b * 4 + h) * 256 + i) * 256 + k] = f2bf(ex[h] / s * g[h]);
    }
}

// ---------------- triplet contraction: per (b,h): O = W(256x256) @ V2(256x4096), in & out ----------------
// Double-buffered: stage kc+1 via GLOBAL_LOAD_ASYNC_TO_LDS_B128 while WMMAs run on stage kc (ASYNCcnt).
__global__ __launch_bounds__(256) void k_trip_gemm(
    const bf16_t* __restrict__ Win, const bf16_t* __restrict__ Wout,
    const bf16_t* __restrict__ V2, bf16_t* __restrict__ Opair)
{
    __shared__ bf16_t Ain[2][128 * 32];
    __shared__ bf16_t Aout[2][128 * 32];
    __shared__ bf16_t Bt[2][128 * 32];   // transposed: [col][k], col = jj*16+d
    const int tid = threadIdx.x, lane = tid & 31, wv = tid >> 5;
    const int bh = blockIdx.x >> 6, rest = blockIdx.x & 63;
    const int ib = rest >> 5, jb = rest & 31;
    const int b = bh >> 2, h = bh & 3;
    const int ibase = ib * 128, jbase = jb * 8;
    const int p = wv >> 2, mloc = wv & 3;             // part: 0=in, 1=out; 2 Mtiles per wave

    // issue one 24KB stage (6 async b128 loads per thread)
    auto issue = [&](int kc, int buf) {
        const int k0 = kc * 32;
#pragma unroll
        for (int t = tid; t < 512; t += 256) {
            const int row = t >> 2, k8 = (t & 3) * 8;
            const long widx = ((long)bh * 256 + ibase + row) * 256 + k0 + k8;
            async_ld16(lds_off32(&Ain[buf][row * 32 + k8]), &Win[widx]);
            async_ld16(lds_off32(&Aout[buf][row * 32 + k8]), &Wout[widx]);
            const int jj = row >> 4, d = row & 15;
            async_ld16(lds_off32(&Bt[buf][row * 32 + k8]),
                       &V2[(((long)bh * 256 + jbase + jj) * 16 + d) * 256 + k0 + k8]);
        }
    };

    issue(0, 0);
    wait_async0();
    __syncthreads();

    v8f acc[2][8] = {};
    for (int kc = 0; kc < 8; ++kc) {
        const int cur = kc & 1;
        if (kc < 7) issue(kc + 1, cur ^ 1);              // DMA overlaps WMMA below
        const bf16_t* Asel = p ? Aout[cur] : Ain[cur];
#pragma unroll
        for (int q = 0; q < 2; ++q)
#pragma unroll
            for (int nt = 0; nt < 8; ++nt)
                acc[q][nt] = wmma_tile(&Asel[(mloc * 2 + q) * 16 * 32], 32, &Bt[cur][nt * 16 * 32], 32,
                                       acc[q][nt], lane);
        if (kc < 7) wait_async0();
        __syncthreads();
    }

    const int n = lane & 15, mh = (lane >> 4) * 8;
#pragma unroll
    for (int q = 0; q < 2; ++q)
#pragma unroll
        for (int nt = 0; nt < 8; ++nt) {
            const int j = jbase + nt;
#pragma unroll
            for (int v = 0; v < 8; ++v) {
                const int iRow = ibase + (mloc * 2 + q) * 16 + mh + v;
                Opair[(((long)b * 256 + iRow) * 256 + j) * 128 + h * 32 + p * 16 + n] = f2bf(acc[q][nt][v]);
            }
        }
}

// ---------------- E_upd_pre = Effn + Opair @ W_top + b_top ; Xln2 = LN_effn(...) (WtopT [64][128]) ----------------
__global__ __launch_bounds__(256) void k_pair_top(
    const bf16_t* __restrict__ Opair, const bf16_t* __restrict__ WtopT,
    const float* __restrict__ btop, const float* __restrict__ Effn,
    float* __restrict__ Eupd, bf16_t* __restrict__ Xln2,
    const float* __restrict__ gamma, const float* __restrict__ beta)
{
    extern __shared__ char smem[];
    bf16_t* Xs = (bf16_t*)smem;              // 128x128
    bf16_t* Ws = (bf16_t*)(smem + 32768);    // 64n x 128k
    float*  Ys = (float*)(smem + 49152);     // 128x64
    const int tid = threadIdx.x, lane = tid & 31, wv = tid >> 5;
    const long mbase = (long)blockIdx.x * 128;
    copy_bf16_v8(Xs, Opair + mbase * 128, 128 * 128, tid, 256);
    copy_bf16_v8(Ws, WtopT, 64 * 128, tid, 256);
    __syncthreads();
    {
        const int mt = wv, n = lane & 15, mh = (lane >> 4) * 8;
#pragma unroll
        for (int nt = 0; nt < 4; ++nt) {
            v8f acc = {};
#pragma unroll
            for (int k0 = 0; k0 < 128; k0 += 32)
                acc = wmma_tile(&Xs[mt * 16 * 128 + k0], 128, &Ws[nt * 16 * 128 + k0], 128, acc, lane);
#pragma unroll
            for (int v = 0; v < 8; ++v) {
                int r = mt * 16 + mh + v, c = nt * 16 + n;
                float y = acc[v] + btop[c] + Effn[(mbase + r) * 64 + c];
                Ys[r * 64 + c] = y;
                Eupd[(mbase + r) * 64 + c] = y;
            }
        }
    }
    __syncthreads();
    if (tid < 128) {
        float m = 0.f;
#pragma unroll
        for (int c = 0; c < 64; ++c) m += Ys[tid * 64 + c];
        m *= (1.0f / 64.0f);
        float var = 0.f;
#pragma unroll
        for (int c = 0; c < 64; ++c) { float d = Ys[tid * 64 + c] - m; var += d * d; }
        var *= (1.0f / 64.0f);
        float rs = rsqrtf(var + 1e-5f);
#pragma unroll
        for (int c = 0; c < 64; ++c)
            Xln2[(mbase + tid) * 64 + c] = f2bf((Ys[tid * 64 + c] - m) * rs * gamma[c] + beta[c]);
    }
}

// ---------------- node: H_attn = H + o@Wo + bo ; Hln = LN_nffn(H_attn) (WoT [128][128]) ----------------
__global__ __launch_bounds__(256) void k_node_out(
    const float* __restrict__ Obuf, const bf16_t* __restrict__ WoT,
    const float* __restrict__ bo, const float* __restrict__ H,
    float* __restrict__ Hattn, bf16_t* __restrict__ Hln,
    const float* __restrict__ gamma, const float* __restrict__ beta)
{
    extern __shared__ char smem[];
    bf16_t* Xs = (bf16_t*)smem;              // 128x128
    bf16_t* Ws = (bf16_t*)(smem + 32768);    // 128n x 128k
    float*  Ys = (float*)(smem + 65536);     // 128x128
    const int tid = threadIdx.x, lane = tid & 31, wv = tid >> 5;
    const long mbase = (long)blockIdx.x * 128;
    for (int i = tid; i < 128 * 128; i += 256) Xs[i] = f2bf(Obuf[mbase * 128 + i]);
    copy_bf16_v8(Ws, WoT, 128 * 128, tid, 256);
    __syncthreads();
    {
        const int mt = wv, n = lane & 15, mh = (lane >> 4) * 8;
        for (int nt = 0; nt < 8; ++nt) {
            v8f acc = {};
#pragma unroll
            for (int k0 = 0; k0 < 128; k0 += 32)
                acc = wmma_tile(&Xs[mt * 16 * 128 + k0], 128, &Ws[nt * 16 * 128 + k0], 128, acc, lane);
#pragma unroll
            for (int v = 0; v < 8; ++v) {
                int r = mt * 16 + mh + v, c = nt * 16 + n;
                float y = acc[v] + bo[c] + H[(mbase + r) * 128 + c];
                Ys[r * 128 + c] = y;
                Hattn[(mbase + r) * 128 + c] = y;
            }
        }
    }
    __syncthreads();
    if (tid < 128) {
        float m = 0.f;
        for (int c = 0; c < 128; ++c) m += Ys[tid * 128 + c];
        m *= (1.0f / 128.0f);
        float var = 0.f;
        for (int c = 0; c < 128; ++c) { float d = Ys[tid * 128 + c] - m; var += d * d; }
        var *= (1.0f / 128.0f);
        float rs = rsqrtf(var + 1e-5f);
        for (int c = 0; c < 128; ++c)
            Hln[(mbase + tid) * 128 + c] = f2bf((Ys[tid * 128 + c] - m) * rs * gamma[c] + beta[c]);
    }
}

// ---------------- node FFN (W1nT [512][128], W2nT [128][512]) ----------------
__global__ __launch_bounds__(256) void k_node_ffn(
    const bf16_t* __restrict__ Xln, const float* __restrict__ Hattn,
    const bf16_t* __restrict__ W1T, const float* __restrict__ b1,
    const bf16_t* __restrict__ W2T, const float* __restrict__ b2,
    float* __restrict__ OutH)
{
    extern __shared__ char smem[];
    bf16_t* Xs  = (bf16_t*)smem;                    // 128x128
    bf16_t* Wc  = (bf16_t*)(smem + 32768);          // 128n x 128k chunk
    bf16_t* Hs  = (bf16_t*)(smem + 65536);          // 128x512
    float*  bsh = (float*)(smem + 65536 + 131072);  // 512
    const int tid = threadIdx.x, lane = tid & 31, wv = tid >> 5;
    const long mbase = (long)blockIdx.x * 128;
    copy_bf16_v8(Xs, Xln + mbase * 128, 128 * 128, tid, 256);
    for (int i = tid; i < 512; i += 256) bsh[i] = b1[i];
    // phase 1: H = gelu(X @ W1n + b1n), hidden in 4 chunks of 128 cols
    for (int hc = 0; hc < 4; ++hc) {
        __syncthreads();
        copy_bf16_v8(Wc, W1T + (long)hc * 128 * 128, 128 * 128, tid, 256);
        __syncthreads();
        const int mt = wv, n = lane & 15, mh = (lane >> 4) * 8;
        for (int nt = 0; nt < 8; ++nt) {
            v8f acc = {};
#pragma unroll
            for (int k0 = 0; k0 < 128; k0 += 32)
                acc = wmma_tile(&Xs[mt * 16 * 128 + k0], 128, &Wc[nt * 16 * 128 + k0], 128, acc, lane);
#pragma unroll
            for (int v = 0; v < 8; ++v)
                Hs[(mt * 16 + mh + v) * 512 + hc * 128 + nt * 16 + n] =
                    f2bf(gelu_exact(acc[v] + bsh[hc * 128 + nt * 16 + n]));
        }
    }
    __syncthreads();
    if (tid < 128) bsh[tid] = b2[tid];
    // phase 2: acc = H @ W2n (K in 4 chunks of 128); W2nT chunk: [128n][128k] slice at kc
    v8f acc[8] = {};
    for (int kc = 0; kc < 4; ++kc) {
        __syncthreads();
        for (int t = tid; t < 2048; t += 256) {          // 128 rows x 16 v8bf
            const int nrow = t >> 4, k8 = (t & 15) * 8;
            ((v8bf*)Wc)[t] = *(const v8bf*)&W2T[nrow * 512 + kc * 128 + k8];
        }
        __syncthreads();
        const int mt = wv;
#pragma unroll
        for (int nt = 0; nt < 8; ++nt)
#pragma unroll
            for (int k0 = 0; k0 < 128; k0 += 32)
                acc[nt] = wmma_tile(&Hs[mt * 16 * 512 + kc * 128 + k0], 512, &Wc[nt * 16 * 128 + k0], 128, acc[nt], lane);
    }
    {
        const int mt = wv, n = lane & 15, mh = (lane >> 4) * 8;
#pragma unroll
        for (int nt = 0; nt < 8; ++nt)
#pragma unroll
            for (int v = 0; v < 8; ++v) {
                int r = mt * 16 + mh + v, c = nt * 16 + n;
                OutH[(mbase + r) * 128 + c] = acc[nt][v] + bsh[c] + 2.0f * Hattn[(mbase + r) * 128 + c];
            }
    }
}

// ---------------- host ----------------
extern "C" void kernel_launch(void* const* d_in, const int* in_sizes, int n_in,
                              void* d_out, int out_size, void* d_ws, size_t ws_size,
                              hipStream_t stream)
{
    (void)in_sizes; (void)n_in; (void)out_size; (void)ws_size;
    const float* H     = (const float*)d_in[0];
    const float* E     = (const float*)d_in[1];
    const float* Wq    = (const float*)d_in[2];
    const float* Wk    = (const float*)d_in[3];
    const float* Wv    = (const float*)d_in[4];
    const float* W_bg  = (const float*)d_in[5];
    const float* Wo    = (const float*)d_in[6];
    const float* bo    = (const float*)d_in[7];
    const float* W_a2e = (const float*)d_in[8];
    const float* b_a2e = (const float*)d_in[9];
    const float* g_ea  = (const float*)d_in[10];
    const float* b_ea  = (const float*)d_in[11];
    const float* g_effn= (const float*)d_in[12];
    const float* b_effn= (const float*)d_in[13];
    const float* W1e   = (const float*)d_in[14];
    const float* b1e   = (const float*)d_in[15];
    const float* W2e   = (const float*)d_in[16];
    const float* b2e   = (const float*)d_in[17];
    const float* g_trip= (const float*)d_in[18];
    const float* b_trip= (const float*)d_in[19];
    const float* Wv_in = (const float*)d_in[20];
    const float* W_gb2 = (const float*)d_in[21];
    const float* W_top = (const float*)d_in[22];
    const float* b_top = (const float*)d_in[23];
    const float* g_nffn= (const float*)d_in[24];
    const float* b_nffn= (const float*)d_in[25];
    const float* W1n   = (const float*)d_in[26];
    const float* b1n   = (const float*)d_in[27];
    const float* W2n   = (const float*)d_in[28];
    const float* b2n   = (const float*)d_in[29];

    char* ws = (char*)d_ws;
    const size_t MBy = 1ull << 20;
    // bf16 weight pool, all TRANSPOSED [N][K]  (<1MB)
    bf16_t* wb     = (bf16_t*)ws;
    bf16_t* Wqkv_b = wb;               // [384][128]
    bf16_t* Wo_b   = wb + 49152;       // [128][128]
    bf16_t* W1e_b  = wb + 65536;       // [256][64]
    bf16_t* W2e_b  = wb + 81920;       // [64][256]
    bf16_t* Wvin_b = wb + 98304;       // [64][64]
    bf16_t* Wtop_b = wb + 102400;      // [64][128]
    bf16_t* W1n_b  = wb + 110592;      // [512][128]
    bf16_t* W2n_b  = wb + 176128;      // [128][512]
    // activations
    float*  QKV   = (float*)(ws + 1 * MBy);                       // 1024x384
    float*  Obuf  = (float*)(ws + 1 * MBy + 1536 * 1024);         // 1024x128
    float*  Hattn = (float*)(ws + 1 * MBy + 2048 * 1024);         // 1024x128
    bf16_t* Hln   = (bf16_t*)(ws + 1 * MBy + 2560 * 1024);        // 1024x128
    float*  gate2 = (float*)(ws + 4 * MBy);                       // ME x 4
    float*  bias2 = (float*)(ws + 8 * MBy);                       // ME x 4
    bf16_t* Win   = (bf16_t*)(ws + 12 * MBy);                     // [b][h][i][k]
    bf16_t* Wout_ = (bf16_t*)(ws + 14 * MBy);                     // [b][h][i][k]
    float*  Eagg  = (float*)(ws + 16 * MBy);                      // ME x 64 (reused: Opair)
    bf16_t* Xln   = (bf16_t*)(ws + 80 * MBy);                     // ME x 64 (reused: Xln2)
    float*  Effn  = (float*)(ws + 112 * MBy);                     // ME x 64
    bf16_t* V2    = (bf16_t*)(ws + 176 * MBy);                    // [b][h][j][d][k] (ME x 64)
    bf16_t* En    = (bf16_t*)(ws + 208 * MBy);                    // ME x 64
    float*  Eupd  = (float*)(ws + 176 * MBy);                     // overlays V2 + En
    bf16_t* Opair = (bf16_t*)(ws + 16 * MBy);                     // overlays Eagg

    float* OutH = (float*)d_out;
    float* OutE = OutH + MN * 128;

    // weights -> bf16 transposed
    k_convert<<<(128*128+255)/256, 256, 0, stream>>>(Wq,    Wqkv_b, 128, 128, 128, 0);
    k_convert<<<(128*128+255)/256, 256, 0, stream>>>(Wk,    Wqkv_b, 128, 128, 128, 128);
    k_convert<<<(128*128+255)/256, 256, 0, stream>>>(Wv,    Wqkv_b, 128, 128, 128, 256);
    k_convert<<<(128*128+255)/256, 256, 0, stream>>>(Wo,    Wo_b,   128, 128, 128, 0);
    k_convert<<<(64*256+255)/256,  256, 0, stream>>>(W1e,   W1e_b,  64, 256, 64, 0);
    k_convert<<<(256*64+255)/256,  256, 0, stream>>>(W2e,   W2e_b,  256, 64, 256, 0);
    k_convert<<<(64*64+255)/256,   256, 0, stream>>>(Wv_in, Wvin_b, 64, 64, 64, 0);
    k_convert<<<(128*64+255)/256,  256, 0, stream>>>(W_top, Wtop_b, 128, 64, 128, 0);
    k_convert<<<(128*512+255)/256, 256, 0, stream>>>(W1n,   W1n_b,  128, 512, 128, 0);
    k_convert<<<(512*128+255)/256, 256, 0, stream>>>(W2n,   W2n_b,  512, 128, 512, 0);

    // node projections + attention + edge stage 1
    k_qkv<<<8, 256, 65536, stream>>>(H, Wqkv_b, QKV);
    k_attn_row<<<1024, 256, 0, stream>>>(E, QKV, W_bg, W_a2e, b_a2e, g_ea, b_ea, g_effn, b_effn,
                                         Obuf, Eagg, Xln);
    // E_ffn = 2*E_agg + FFN(LN(E_agg)) ; En = LN_trip(E_ffn)
    edge_ffn_kernel<<<2048, 256, 115712, stream>>>(Xln, Eagg, W1e_b, b1e, W2e_b, b2e,
                                                   Effn, En, g_trip, b_trip, 1);
    // triplet path
    k_trip_prep<<<2048, 256, 0, stream>>>(En, Wvin_b, W_gb2, V2, gate2, bias2);
    k_softmax_in<<<1024, 256, 0, stream>>>(bias2, gate2, Win);
    k_softmax_out<<<1024, 256, 0, stream>>>(bias2, gate2, Wout_);
    k_trip_gemm<<<1024, 256, 0, stream>>>(Win, Wout_, V2, Opair);
    // E_upd_pre = E_ffn + o_pair@W_top + b_top ; Xln2 = LN_effn(E_upd_pre)
    k_pair_top<<<2048, 256, 81920, stream>>>(Opair, Wtop_b, b_top, Effn, Eupd, Xln, g_effn, b_effn);
    // final: E_upd = 2*E_upd_pre + FFN(LN(E_upd_pre))
    edge_ffn_kernel<<<2048, 256, 115712, stream>>>(Xln, Eupd, W1e_b, b1e, W2e_b, b2e,
                                                   OutE, (bf16_t*)0, g_trip, b_trip, 0);
    // node tail
    k_node_out<<<8, 256, 131072, stream>>>(Obuf, Wo_b, bo, H, Hattn, Hln, g_nffn, b_nffn);
    k_node_ffn<<<8, 256, 198656, stream>>>(Hln, Hattn, W1n_b, b1n, W2n_b, b2n, OutH);
}
